// RNNCCModel_30889404793452
// MI455X (gfx1250) — matrix operand order repaired and checked by reading, not compiled, not run on hardware
//
#include <hip/hip_runtime.h>
#include <math.h>

#define B_ 64
#define S_ 1024
#define D_ 1024
#define C_ 10
#define N3D (3 * D_)

typedef __attribute__((ext_vector_type(16))) __bf16 v16bf;
typedef __attribute__((ext_vector_type(8)))  __bf16 v8bf;
typedef __attribute__((ext_vector_type(8)))  float  v8f;

__device__ __forceinline__ float wave_sum(float v) {
#pragma unroll
    for (int off = 16; off > 0; off >>= 1) v += __shfl_xor(v, off, 32);
    return v;
}
__device__ __forceinline__ float wave_max(float v) {
#pragma unroll
    for (int off = 16; off > 0; off >>= 1) v = fmaxf(v, __shfl_xor(v, off, 32));
    return v;
}

// ---------------- init: zero h (f32 + bf16) and score chain ----------------
__global__ void init_zero(float* __restrict__ h, __bf16* __restrict__ h_bf,
                          float* __restrict__ score) {
    int i = blockIdx.x * blockDim.x + threadIdx.x;
    if (i < B_ * D_) { h[i] = 0.f; h_bf[i] = (__bf16)0.f; }
    if (i < B_) score[i] = 0.f;
}

// ---------------- one-time: f32 weights -> bf16, packed [rows, D_] row-major ----------------
__global__ void cvt_weights(const float* __restrict__ W, int ldw,
                            __bf16* __restrict__ out, int total) {
    int i = blockIdx.x * blockDim.x + threadIdx.x;
    if (i < total) {
        int r = i >> 10;            // / D_
        int c = i & (D_ - 1);       // % D_
        out[i] = (__bf16)W[(size_t)r * ldw + c];
    }
}

// ---------------- phase A: logits[b,s] = feat[b,s,:] . a_w ----------------
// one wave per (b,s) row; fully coalesced float4 loads
__global__ void logits_kernel(const float* __restrict__ feat,
                              const float* __restrict__ a_w,
                              float* __restrict__ logits) {
    const int wid  = threadIdx.x >> 5;
    const int lane = threadIdx.x & 31;
    const int gw   = blockIdx.x * (blockDim.x >> 5) + wid;  // (b*S + s)
    const float4* fr = (const float4*)(feat + (size_t)gw * D_);
    const float4* ar = (const float4*)a_w;
    float acc = 0.f;
#pragma unroll
    for (int i = 0; i < 8; ++i) {
        float4 f = fr[lane + 32 * i];
        float4 a = ar[lane + 32 * i];
        acc += f.x * a.x + f.y * a.y + f.z * a.z + f.w * a.w;
    }
    acc = wave_sum(acc);
    if (lane == 0) logits[gw] = acc;
}

// ---------------- phase B: softmax over S, in place ----------------
__global__ void softmax_kernel(float* __restrict__ logits) {
    __shared__ float red[32];
    const int b = blockIdx.x, t = threadIdx.x;  // 1024 threads
    float l = logits[b * S_ + t];
    float m = wave_max(l);
    if ((t & 31) == 0) red[t >> 5] = m;
    __syncthreads();
    float gmax = -1e30f;
#pragma unroll
    for (int i = 0; i < 32; ++i) gmax = fmaxf(gmax, red[i]);
    __syncthreads();
    float e = expf(l - gmax);
    float s = wave_sum(e);
    if ((t & 31) == 0) red[t >> 5] = s;
    __syncthreads();
    float gsum = 0.f;
#pragma unroll
    for (int i = 0; i < 32; ++i) gsum += red[i];
    logits[b * S_ + t] = e / gsum;
}

// ---------------- phase C: pooled_bf[b,d] = (bf16) sum_s w[b,s]*feat[b,s,d] ----------------
__global__ void pooled_kernel(const float* __restrict__ w,
                              const float* __restrict__ feat,
                              __bf16* __restrict__ pooled_bf) {
    __shared__ float ws[S_];
    const int b = blockIdx.x, t = threadIdx.x;  // 1024 threads, t = d
    ws[t] = w[b * S_ + t];
    __syncthreads();
    const float* fb = feat + (size_t)b * S_ * D_ + t;
    float acc = 0.f;
#pragma unroll 8
    for (int s = 0; s < S_; ++s) acc += ws[s] * fb[(size_t)s * D_];
    pooled_bf[b * D_ + t] = (__bf16)acc;
}

// ---------------- extract last column of W_ih (the score column) ----------------
__global__ void extract_wihd(const float* __restrict__ W_ih, float* __restrict__ wihd) {
    int i = blockIdx.x * blockDim.x + threadIdx.x;
    if (i < N3D) wihd[i] = W_ih[(size_t)i * (D_ + 1) + D_];
}

// ---------------- WMMA GEMM: out[64, N3D] = A[64,D] * W[N3D,D]^T + bias ----------------
// bf16 inputs pre-packed row-major; one wave computes TWO 16x16 tiles along N
// (A fragment reused across two independent WMMA accumulator chains).
// Fragment layouts used here (wave32; contiguous K, not the ISA table order):
//   A 16x32 bf16: lane = m + 16*h; elem j   -> K = 16h+j   (16 contiguous bf16)
//   B 32x16 bf16: lane = n + 16*h; elem j   -> K = 16h+j   (16 contiguous bf16)
//   C 16x16 f32 : vgpr r, lane = n + 16*h   -> row m = r + 8h, col n
__global__ void gemm_wmma_bf16(const __bf16* __restrict__ A,   // [64, D_]
                               const __bf16* __restrict__ W,   // [N3D, D_]
                               const float* __restrict__ bias,
                               float* __restrict__ out) {
    const int lane = threadIdx.x & 31;
    const int hh   = lane >> 4;
    const int ln   = lane & 15;
    const int n0   = blockIdx.x * 32;    // two N tiles: n0, n0+16
    const int m0   = blockIdx.y * 16;

    v8f c0 = {}, c1 = {};
    const __bf16* arow  = A + (size_t)(m0 + ln) * D_ + 16 * hh;
    const __bf16* wrow0 = W + (size_t)(n0 + ln) * D_ + 16 * hh;
    const __bf16* wrow1 = wrow0 + (size_t)16 * D_;

#pragma unroll 2
    for (int k0 = 0; k0 < D_; k0 += 32) {
        v8bf a_lo = *(const v8bf*)(arow + k0);        // K = 16h + 0..7
        v8bf a_hi = *(const v8bf*)(arow + k0 + 8);    // K = 16h + 8..15
        v16bf a = __builtin_shufflevector(a_lo, a_hi,
                     0, 1, 2, 3, 4, 5, 6, 7, 8, 9, 10, 11, 12, 13, 14, 15);
        v16bf b0 = *(const v16bf*)(wrow0 + k0);       // K = 16h + 0..15
        v16bf b1 = *(const v16bf*)(wrow1 + k0);
        c0 = __builtin_amdgcn_wmma_f32_16x16x32_bf16(false, a, false, b0,
                                                     (short)0, c0, false, false);
        c1 = __builtin_amdgcn_wmma_f32_16x16x32_bf16(false, a, false, b1,
                                                     (short)0, c1, false, false);
    }
    const float bb0 = bias[n0 + ln];
    const float bb1 = bias[n0 + 16 + ln];
#pragma unroll
    for (int r = 0; r < 8; ++r) {
        float* orow = out + (size_t)(m0 + r + 8 * hh) * N3D;
        orow[n0 + ln]      = c0[r] + bb0;
        orow[n0 + 16 + ln] = c1[r] + bb1;
    }
}

// ---------------- per-step gates + in-place h update + score dot ----------------
__global__ void gru_gate_kernel(const float* __restrict__ gi_base,
                                const float* __restrict__ gh,
                                const float* __restrict__ wihd,
                                const float* __restrict__ W_out,
                                const float* __restrict__ b_out,
                                float* __restrict__ h,
                                __bf16* __restrict__ h_bf,
                                float* __restrict__ score,
                                float* __restrict__ out, int step) {
    __shared__ float red[8];
    __shared__ float sprev_sh;
    const int b = blockIdx.x, t = threadIdx.x;  // 256 threads, 4 d each
    if (t == 0) sprev_sh = score[b];
    __syncthreads();
    const float sprev = sprev_sh;
    const float* gib = gi_base + (size_t)b * N3D;
    const float* ghb = gh + (size_t)b * N3D;
    float* hb = h + (size_t)b * D_;
    __bf16* hbb = h_bf + (size_t)b * D_;
    float partial = 0.f;
#pragma unroll
    for (int i = 0; i < 4; ++i) {
        const int d = t + 256 * i;
        float i_r = gib[d]          + sprev * wihd[d];
        float i_z = gib[D_ + d]     + sprev * wihd[D_ + d];
        float i_n = gib[2 * D_ + d] + sprev * wihd[2 * D_ + d];
        float h_r = ghb[d], h_z = ghb[D_ + d], h_n = ghb[2 * D_ + d];
        float r = 1.f / (1.f + expf(-(i_r + h_r)));
        float z = 1.f / (1.f + expf(-(i_z + h_z)));
        float n = tanhf(i_n + r * h_n);
        float hp = hb[d];
        float hn = (1.f - z) * n + z * hp;
        hb[d] = hn;
        hbb[d] = (__bf16)hn;
        partial += hn * W_out[d];
    }
    partial = wave_sum(partial);
    if ((t & 31) == 0) red[t >> 5] = partial;
    __syncthreads();
    if (t == 0) {
        float tot = 0.f;
#pragma unroll
        for (int i = 0; i < 8; ++i) tot += red[i];
        float sn = 1.f / (1.f + expf(-(tot + b_out[0])));
        score[b] = sn;
        out[b * C_ + step] = sn;
    }
}

extern "C" void kernel_launch(void* const* d_in, const int* in_sizes, int n_in,
                              void* d_out, int out_size, void* d_ws, size_t ws_size,
                              hipStream_t stream) {
    const float* feat  = (const float*)d_in[0];
    const float* a_w   = (const float*)d_in[1];
    const float* W_ih  = (const float*)d_in[2];
    const float* W_hh  = (const float*)d_in[3];
    const float* b_ih  = (const float*)d_in[4];
    const float* b_hh  = (const float*)d_in[5];
    const float* W_out = (const float*)d_in[6];
    const float* b_out = (const float*)d_in[7];
    float* out = (float*)d_out;

    // ---- workspace carve-up: f32 regions first, then bf16 regions ----
    float* wsf = (float*)d_ws;
    float* logits  = wsf;  wsf += B_ * S_;     // softmax weights in place
    float* gi_base = wsf;  wsf += B_ * N3D;
    float* gh      = wsf;  wsf += B_ * N3D;
    float* h       = wsf;  wsf += B_ * D_;
    float* wihd    = wsf;  wsf += N3D;
    float* score   = wsf;  wsf += B_;
    __bf16* wsb = (__bf16*)wsf;
    __bf16* pooled_bf = wsb;  wsb += B_ * D_;
    __bf16* h_bf      = wsb;  wsb += B_ * D_;
    __bf16* wih_bf    = wsb;  wsb += (size_t)N3D * D_;
    __bf16* whh_bf    = wsb;  wsb += (size_t)N3D * D_;

    // ---- one-time setup ----
    init_zero<<<(B_ * D_ + 255) / 256, 256, 0, stream>>>(h, h_bf, score);
    cvt_weights<<<(N3D * D_ + 255) / 256, 256, 0, stream>>>(W_ih, D_ + 1, wih_bf, N3D * D_);
    cvt_weights<<<(N3D * D_ + 255) / 256, 256, 0, stream>>>(W_hh, D_,     whh_bf, N3D * D_);
    extract_wihd<<<(N3D + 255) / 256, 256, 0, stream>>>(W_ih, wihd);

    // ---- attention pooling (bandwidth-bound, fp32) ----
    logits_kernel<<<(B_ * S_) / 8, 256, 0, stream>>>(feat, a_w, logits);
    softmax_kernel<<<B_, S_, 0, stream>>>(logits);
    pooled_kernel<<<B_, D_, 0, stream>>>(logits, feat, pooled_bf);

    // ---- GRU chain ----
    dim3 gg(N3D / 32, B_ / 16);   // 96 x 4 blocks, 1 wave each, 2 N-tiles per wave
    // gi_base = pooled @ W_ih[:, :D].T + b_ih  (once; score column is a per-step rank-1 fixup)
    gemm_wmma_bf16<<<gg, 32, 0, stream>>>(pooled_bf, wih_bf, b_ih, gi_base);

    for (int c = 0; c < C_; ++c) {
        // gh = h @ W_hh.T + b_hh
        gemm_wmma_bf16<<<gg, 32, 0, stream>>>(h_bf, whh_bf, b_hh, gh);
        gru_gate_kernel<<<B_, 256, 0, stream>>>(gi_base, gh, wihd, W_out, b_out,
                                                h, h_bf, score, out, c);
    }
}
